// EncoderBlock_38783554682883
// MI455X (gfx1250) — matrix operand order repaired and checked by reading, not compiled, not run on hardware
//
#include <hip/hip_runtime.h>

// ---------------------------------------------------------------------------
// Transformer encoder block for MI455X (gfx1250, wave32, WMMA).
// - All matmuls: v_wmma_f32_16x16x32_bf16 (bf16 in, f32 accumulate).
// - Weights pre-swizzled once into WMMA B-fragment tile order, so every GEMM
//   tile (A raw, B fragment-order) stages as contiguous async DMA
//   (global_load_async_to_lds_b128, ASYNCcnt) with LDS double buffering.
// - Flash attention: score matrix never hits HBM; K/V are produced by the
//   QKV GEMM epilogue directly in fragment order -> bulk async staging.
// ---------------------------------------------------------------------------

typedef __attribute__((ext_vector_type(16))) __bf16 v16bf;
typedef __attribute__((ext_vector_type(8)))  float  v8f;

#define D_MODEL 512
#define SEQ     4096
#define BATCH   2
#define HEADS   8
#define DKH     64
#define FFH     1024
#define MROWS   (BATCH * SEQ)   // 8192
#define NKBLK   (SEQ / 32)      // 128 key blocks per head

#define WMMA_BF16(A_, B_, C_) \
  __builtin_amdgcn_wmma_f32_16x16x32_bf16(false, (A_), false, (B_), (short)0, (C_), false, false)

__device__ __forceinline__ __bf16 f2bf(float f) {
  union { float f; unsigned u; } in; in.f = f;
  unsigned u = in.u;
  unsigned r = u + 0x7fffu + ((u >> 16) & 1u);   // round-to-nearest-even
  union { unsigned short s; __bf16 b; } out;
  out.s = (unsigned short)(r >> 16);
  return out.b;
}

__device__ __forceinline__ unsigned lds_off(const void* p) {
  // Generic pointers to LDS carry the LDS byte offset in the low 32 bits.
  return (unsigned)(unsigned long long)p;
}

// Async bulk copy global -> LDS, 16B per lane, tracked with ASYNCcnt.
__device__ __forceinline__ void async_ld_b128(unsigned ldsoff, const void* gsrc) {
  asm volatile("global_load_async_to_lds_b128 %0, %1, off"
               :: "v"(ldsoff), "v"((unsigned long long)gsrc)
               : "memory");
}
__device__ __forceinline__ void wait_async() {
  asm volatile("s_wait_asynccnt 0x0" ::: "memory");
}

// ---------------------------------------------------------------------------
// f32 -> bf16 conversion (grid-stride), row-major preserved
// ---------------------------------------------------------------------------
__global__ void cvt_f32_bf16(const float* __restrict__ in,
                             __bf16* __restrict__ out, int n) {
  int i = blockIdx.x * blockDim.x + threadIdx.x;
  int stride = gridDim.x * blockDim.x;
  for (; i < n; i += stride) out[i] = f2bf(in[i]);
}

// ---------------------------------------------------------------------------
// Weight conversion + swizzle into WMMA B-fragment tile order:
//   out[ kblk ][ nblk ][ nt ][ lane ][ elem ]  (tile = 32x64 -> 4KB block)
// lane = (n&15) + 16*((k>>4)&1), elem = k&15, nt = (n&63)>>4.
// ---------------------------------------------------------------------------
__global__ void cvt_swz_w(const float* __restrict__ in, __bf16* __restrict__ out,
                          int total, int nShift) {
  int idx = blockIdx.x * blockDim.x + threadIdx.x;
  int stride = gridDim.x * blockDim.x;
  const int nMask = (1 << nShift) - 1;
  const int nBlk  = 1 << (nShift - 6);
  for (; idx < total; idx += stride) {
    int k = idx >> nShift, n = idx & nMask;
    int kblk = k >> 5, nblk = n >> 6;
    int nt = (n & 63) >> 4;
    int la = (n & 15) + ((k >> 4) & 1) * 16;
    int el = k & 15;
    size_t o = ((((size_t)kblk * nBlk + nblk) * 4 + nt) * 32 + la) * 16 + el;
    out[o] = f2bf(in[idx]);
  }
}

// ---------------------------------------------------------------------------
// Tiled bf16 GEMM:  C[M,N] = A[M,K] @ B[K,N] (+bias)(+relu)(+resid f32)
// Block tile 64x64, 8 waves, 2 WMMA accumulators/wave, K-step 32.
// A raw row-major, Bf pre-swizzled fragment order: both tiles staged with one
// async b128 per thread into double-buffered LDS; DMA overlaps the WMMAs.
// MODE: 0 = row-major out, 1 = Q [B*H,S,dk], 2 = K-frag, 3 = V-frag.
// ---------------------------------------------------------------------------
template <int MODE>
__global__ __launch_bounds__(256)
void gemm_bf16(const __bf16* __restrict__ A, const __bf16* __restrict__ Bf,
               const float* __restrict__ bias, const float* __restrict__ resid,
               float* __restrict__ outF, __bf16* __restrict__ outB,
               int N, int K, int relu)
{
  __shared__ __bf16 lsA[2][64 * 32];     // raw A tiles (4KB each)
  __shared__ __bf16 lsB[2][4 * 32 * 16]; // B fragment tiles (4KB each)

  const int t      = threadIdx.x;
  const int lane   = t & 31;
  const int w      = t >> 5;
  const int lane16 = lane & 15;
  const int khalf  = lane >> 4;
  const int wm     = w >> 1;
  const int wn     = w & 1;
  const int mBase  = blockIdx.y * 64;
  const int nBase  = blockIdx.x * 64;

  const int ar = t >> 2, ac = (t & 3) * 8;
  const unsigned aL = lds_off(&lsA[0][0]) + (unsigned)(ar * 64 + ac * 2);
  const unsigned bL = lds_off(&lsB[0][0]) + (unsigned)(t * 16);
  const int bStride = (N >> 6) * 2048;                     // elems per k-block
  const __bf16* aCur = A + (size_t)(mBase + ar) * K + ac;
  const __bf16* bCur = Bf + (size_t)(nBase >> 6) * 2048 + t * 8;

  // prologue: stage tile 0 into buffer 0
  async_ld_b128(aL, aCur);
  async_ld_b128(bL, bCur);
  wait_async();
  __syncthreads();

  v8f acc[2] = {};
  int buf = 0;
  for (int kb = 0; kb < K; kb += 32) {
    const __bf16* aNxt = aCur + 32;
    const __bf16* bNxt = bCur + bStride;
    if (kb + 32 < K) {                       // stage next tile (async DMA)
      async_ld_b128(aL + (buf ^ 1) * 4096u, aNxt);
      async_ld_b128(bL + (buf ^ 1) * 4096u, bNxt);
      if (kb + 64 < K) {                     // and prefetch the one after
        __builtin_prefetch(aNxt + 32, 0, 1);
        __builtin_prefetch(bNxt + bStride, 0, 1);
      }
    }

    v16bf a, b0, b1;
    const __bf16* pa = &lsA[buf][(wm * 16 + lane16) * 32 + 16 * khalf];
#pragma unroll
    for (int i = 0; i < 8; ++i) { a[i] = pa[i]; a[i + 8] = pa[8 + i]; }
    const __bf16* pb = &lsB[buf][((wn * 2) * 32 + lane) * 16];
#pragma unroll
    for (int i = 0; i < 16; ++i) { b0[i] = pb[i]; b1[i] = pb[512 + i]; }
    acc[0] = WMMA_BF16(a, b0, acc[0]);
    acc[1] = WMMA_BF16(a, b1, acc[1]);

    wait_async();
    __syncthreads();
    buf ^= 1;
    aCur = aNxt;
    bCur = bNxt;
  }

#pragma unroll
  for (int ii = 0; ii < 2; ++ii) {
    const int n = nBase + wn * 32 + ii * 16 + lane16;
    const float bs = bias ? bias[n] : 0.0f;
#pragma unroll
    for (int r = 0; r < 8; ++r) {
      int m = mBase + wm * 16 + r + 8 * khalf;
      float v = acc[ii][r] + bs;
      if (relu) v = fmaxf(v, 0.0f);
      if (resid) v += resid[(size_t)m * N + n];
      if (outF) outF[(size_t)m * N + n] = v;
      if (outB) {
        size_t idx;
        if (MODE == 0) {
          idx = (size_t)m * N + n;
        } else {
          int bb = m >> 12, s = m & 4095;          // S = 4096
          int head = n >> 6, ch = n & 63;          // dk = 64
          int bh = bb * HEADS + head;
          if (MODE == 1) {                          // Q: [B*H, S, dk]
            idx = ((size_t)bh * SEQ + s) * 64 + ch;
          } else if (MODE == 2) {                   // K: WMMA B-frag order
            int keyblk = s >> 5, k32 = s & 31;
            int nt = k32 >> 4, kcb = ch >> 5;
            int la = (k32 & 15) + ((ch >> 4) & 1) * 16;
            idx = ((((size_t)bh * NKBLK + keyblk) * 4 + nt * 2 + kcb) * 32 + la) * 16
                  + (ch & 15);
          } else {                                  // V: WMMA B-frag order
            int keyblk = s >> 5, k32 = s & 31;
            int nt = ch >> 4;
            int la = (ch & 15) + (k32 >> 4) * 16;
            idx = ((((size_t)bh * NKBLK + keyblk) * 4 + nt) * 32 + la) * 16
                  + (k32 & 15);
          }
        }
        outB[idx] = f2bf(v);
      }
    }
  }
}

// ---------------------------------------------------------------------------
// Flash attention. Q in [B*H,S,64]; K/V pre-swizzled to fragment order so
// each 32-key tile is a contiguous 4KB block. Double-buffered async staging.
// One block = 128 query rows of one (b,h); 8 waves x 16 rows each.
// ---------------------------------------------------------------------------
__global__ __launch_bounds__(256)
void flash_attn(const __bf16* __restrict__ Q, const __bf16* __restrict__ Kf,
                const __bf16* __restrict__ Vf, __bf16* __restrict__ out)
{
  __shared__ __bf16 sKf[2][2048];     // [nt 0..1][kcb 0..1][lane][16]
  __shared__ __bf16 sVf[2][2048];     // [nt 0..3][lane][16]
  __shared__ float  sS[8][512];       // per-wave scores (f32)
  __shared__ __bf16 sP[8][512];       // per-wave probabilities (row-major)
  __shared__ float  sA[8][16];        // per-row rescale
  __shared__ float  sL[8][16];        // per-row 1/l

  const int t      = threadIdx.x;
  const int lane   = t & 31;
  const int w      = t >> 5;
  const int lane16 = lane & 15;
  const int khalf  = lane >> 4;
  const int bh     = blockIdx.y;
  const int b      = bh >> 3, h = bh & 7;
  const int qrow   = blockIdx.x * 128 + w * 16 + lane16;
  const __bf16* qp = Q + ((size_t)bh * SEQ + qrow) * 64 + 16 * khalf;

  const unsigned kL = lds_off(&sKf[0][0]) + (unsigned)t * 16u;
  const unsigned vL = lds_off(&sVf[0][0]) + (unsigned)t * 16u;
  const __bf16* kCur = Kf + (size_t)bh * NKBLK * 2048 + t * 8;
  const __bf16* vCur = Vf + (size_t)bh * NKBLK * 2048 + t * 8;

  // Q A-fragments for contraction halves [0,32) and [32,64): contiguous b128s
  v16bf a0, a1;
#pragma unroll
  for (int i = 0; i < 8; ++i) {
    a0[i] = qp[i];      a0[i + 8] = qp[8 + i];
    a1[i] = qp[32 + i]; a1[i + 8] = qp[40 + i];
  }

  // prologue: stage key-block 0 into buffer 0
  async_ld_b128(kL, kCur);
  async_ld_b128(vL, vCur);
  wait_async();
  __syncthreads();

  v8f oacc[4] = {};
  float mrow = -1e30f, lrow = 0.0f;
  const float scale = 0.125f;         // 1/sqrt(64)
  int buf = 0;

  for (int kb = 0; kb < NKBLK; ++kb) {
    const __bf16* kNxt = kCur + 2048;
    const __bf16* vNxt = vCur + 2048;
    if (kb + 1 < NKBLK) {                    // async-stage next key block
      async_ld_b128(kL + (buf ^ 1) * 4096u, kNxt);
      async_ld_b128(vL + (buf ^ 1) * 4096u, vNxt);
      if (kb + 2 < NKBLK) {
        __builtin_prefetch(kNxt + 2048, 0, 1);
        __builtin_prefetch(vNxt + 2048, 0, 1);
      }
    }

    // scores: S16x32 = Q(16x64) . K^T(64x32), two 16-key subtiles
#pragma unroll
    for (int nt = 0; nt < 2; ++nt) {
      v8f s = {};
      v16bf b0, b1;
      const __bf16* pk = &sKf[buf][((nt * 2) * 32 + lane) * 16];
#pragma unroll
      for (int i = 0; i < 16; ++i) { b0[i] = pk[i]; b1[i] = pk[512 + i]; }
      s = WMMA_BF16(a0, b0, s);
      s = WMMA_BF16(a1, b1, s);
#pragma unroll
      for (int r = 0; r < 8; ++r)
        sS[w][(r + 8 * khalf) * 32 + nt * 16 + lane16] = s[r] * scale;
    }

    // online softmax update: lane j (<16) owns row j (per-wave LDS, in-order)
    if (lane < 16) {
      float sc[32];
#pragma unroll
      for (int j = 0; j < 32; ++j) sc[j] = sS[w][lane * 32 + j];
      float mx = mrow;
#pragma unroll
      for (int j = 0; j < 32; ++j) mx = fmaxf(mx, sc[j]);
      float ladd = 0.0f;
#pragma unroll
      for (int j = 0; j < 32; ++j) {
        float p = __expf(sc[j] - mx);
        sP[w][lane * 32 + j] = f2bf(p);
        ladd += p;
      }
      float alpha = __expf(mrow - mx);
      lrow = lrow * alpha + ladd;
      mrow = mx;
      sA[w][lane] = alpha;
    }
#pragma unroll
    for (int r = 0; r < 8; ++r) {
      float al = sA[w][r + 8 * khalf];
#pragma unroll
      for (int nt = 0; nt < 4; ++nt) oacc[nt][r] *= al;
    }

    // O += P(16x32) . V(32x64)
    v16bf pa;
    const __bf16* pp = &sP[w][lane16 * 32 + 16 * khalf];
#pragma unroll
    for (int i = 0; i < 8; ++i) { pa[i] = pp[i]; pa[i + 8] = pp[8 + i]; }
#pragma unroll
    for (int nt = 0; nt < 4; ++nt) {
      v16bf bv;
      const __bf16* pv = &sVf[buf][(nt * 32 + lane) * 16];
#pragma unroll
      for (int i = 0; i < 16; ++i) bv[i] = pv[i];
      oacc[nt] = WMMA_BF16(pa, bv, oacc[nt]);
    }

    wait_async();
    __syncthreads();
    buf ^= 1;
    kCur = kNxt;
    vCur = vNxt;
  }

  if (lane < 16) sL[w][lane] = 1.0f / lrow;
  const int srowBase = blockIdx.x * 128 + w * 16;
#pragma unroll
  for (int r = 0; r < 8; ++r) {
    float iv = sL[w][r + 8 * khalf];
    int sg = srowBase + r + 8 * khalf;
    size_t o = ((size_t)(b * SEQ + sg)) * D_MODEL + h * DKH + lane16;
    out[o +  0] = f2bf(oacc[0][r] * iv);
    out[o + 16] = f2bf(oacc[1][r] * iv);
    out[o + 32] = f2bf(oacc[2][r] * iv);
    out[o + 48] = f2bf(oacc[3][r] * iv);
  }
}

// ---------------------------------------------------------------------------
// LayerNorm matching reference: unbiased std (ddof=1), divide by (std + eps).
// ---------------------------------------------------------------------------
__global__ __launch_bounds__(256)
void layernorm(const float* __restrict__ in, const float* __restrict__ g,
               const float* __restrict__ be, float* __restrict__ outF,
               __bf16* __restrict__ outB)
{
  __shared__ float red[256];
  const int row = blockIdx.x, t = threadIdx.x;
  const float* x = in + (size_t)row * D_MODEL;
  float v0 = x[t], v1 = x[t + 256];

  red[t] = v0 + v1;
  __syncthreads();
  for (int off = 128; off > 0; off >>= 1) {
    if (t < off) red[t] += red[t + off];
    __syncthreads();
  }
  float mean = red[0] * (1.0f / 512.0f);
  __syncthreads();

  float d0 = v0 - mean, d1 = v1 - mean;
  red[t] = d0 * d0 + d1 * d1;
  __syncthreads();
  for (int off = 128; off > 0; off >>= 1) {
    if (t < off) red[t] += red[t + off];
    __syncthreads();
  }
  float var = red[0] * (1.0f / 511.0f);     // ddof = 1
  float inv = 1.0f / (sqrtf(var) + 1e-6f);  // (std + eps)

  float r0 = g[t] * (d0 * inv) + be[t];
  float r1 = g[t + 256] * (d1 * inv) + be[t + 256];
  if (outF) {
    outF[(size_t)row * D_MODEL + t] = r0;
    outF[(size_t)row * D_MODEL + t + 256] = r1;
  }
  if (outB) {
    outB[(size_t)row * D_MODEL + t] = f2bf(r0);
    outB[(size_t)row * D_MODEL + t + 256] = f2bf(r1);
  }
}

// ---------------------------------------------------------------------------
extern "C" void kernel_launch(void* const* d_in, const int* in_sizes, int n_in,
                              void* d_out, int out_size, void* d_ws, size_t ws_size,
                              hipStream_t stream)
{
  const float* x  = (const float*)d_in[0];
  const float* Wq = (const float*)d_in[1];
  const float* bq = (const float*)d_in[2];
  const float* Wk = (const float*)d_in[3];
  const float* bk = (const float*)d_in[4];
  const float* Wv = (const float*)d_in[5];
  const float* bv = (const float*)d_in[6];
  const float* Wo = (const float*)d_in[7];
  const float* bo = (const float*)d_in[8];
  const float* a1 = (const float*)d_in[9];
  const float* b1 = (const float*)d_in[10];
  const float* a2 = (const float*)d_in[11];
  const float* b2 = (const float*)d_in[12];
  const float* W1 = (const float*)d_in[13];
  const float* c1 = (const float*)d_in[14];
  const float* W2 = (const float*)d_in[15];
  const float* c2 = (const float*)d_in[16];

  char* ws = (char*)d_ws;
  size_t off = 0;
  auto alloc = [&](size_t bytes) -> char* {
    char* p = ws + off;
    off += (bytes + 255) & ~(size_t)255;
    return p;
  };

  __bf16* xb   = (__bf16*)alloc((size_t)MROWS * D_MODEL * 2);
  __bf16* wqb  = (__bf16*)alloc((size_t)D_MODEL * D_MODEL * 2);  // swizzled
  __bf16* wkb  = (__bf16*)alloc((size_t)D_MODEL * D_MODEL * 2);
  __bf16* wvb  = (__bf16*)alloc((size_t)D_MODEL * D_MODEL * 2);
  __bf16* wob  = (__bf16*)alloc((size_t)D_MODEL * D_MODEL * 2);
  __bf16* w1b  = (__bf16*)alloc((size_t)D_MODEL * FFH * 2);
  __bf16* w2b  = (__bf16*)alloc((size_t)FFH * D_MODEL * 2);
  __bf16* qb   = (__bf16*)alloc((size_t)MROWS * D_MODEL * 2);   // Q [B*H,S,dk]
  __bf16* kfr  = (__bf16*)alloc((size_t)MROWS * D_MODEL * 2);   // K frag order
  __bf16* vfr  = (__bf16*)alloc((size_t)MROWS * D_MODEL * 2);   // V frag order
  float*  y1   = (float*)alloc((size_t)MROWS * D_MODEL * 4);
  float*  x1   = (float*)alloc((size_t)MROWS * D_MODEL * 4);
  // Aliases (lifetimes disjoint):
  __bf16* attnb = xb;    // attention concat out; xb dead after QKV GEMMs
  __bf16* hb    = qb;    // FFN hidden (spans qb+kfr, 16MB); dead after attn
  __bf16* x1b   = vfr;   // ln1 bf16 out; vfr dead after attention
  float*  y2    = y1;    // second residual sum; y1 dead after ln1

  // 1) activations to bf16 (row-major); weights to bf16 + fragment swizzle
  cvt_f32_bf16<<<2048, 256, 0, stream>>>(x, xb, MROWS * D_MODEL);
  cvt_swz_w<<<512,  256, 0, stream>>>(Wq, wqb, D_MODEL * D_MODEL, 9);
  cvt_swz_w<<<512,  256, 0, stream>>>(Wk, wkb, D_MODEL * D_MODEL, 9);
  cvt_swz_w<<<512,  256, 0, stream>>>(Wv, wvb, D_MODEL * D_MODEL, 9);
  cvt_swz_w<<<512,  256, 0, stream>>>(Wo, wob, D_MODEL * D_MODEL, 9);
  cvt_swz_w<<<1024, 256, 0, stream>>>(W1, w1b, D_MODEL * FFH, 10);
  cvt_swz_w<<<1024, 256, 0, stream>>>(W2, w2b, FFH * D_MODEL, 9);

  // 2) QKV projections (Q plain head layout, K/V in WMMA fragment order)
  dim3 gQKV(D_MODEL / 64, MROWS / 64);
  gemm_bf16<1><<<gQKV, 256, 0, stream>>>(xb, wqb, bq, nullptr, nullptr, qb,
                                         D_MODEL, D_MODEL, 0);
  gemm_bf16<2><<<gQKV, 256, 0, stream>>>(xb, wkb, bk, nullptr, nullptr, kfr,
                                         D_MODEL, D_MODEL, 0);
  gemm_bf16<3><<<gQKV, 256, 0, stream>>>(xb, wvb, bv, nullptr, nullptr, vfr,
                                         D_MODEL, D_MODEL, 0);

  // 3) flash attention -> concat bf16
  flash_attn<<<dim3(SEQ / 128, BATCH * HEADS), 256, 0, stream>>>(qb, kfr, vfr,
                                                                 attnb);

  // 4) output projection + residual(x) -> y1 (f32)
  gemm_bf16<0><<<dim3(D_MODEL / 64, MROWS / 64), 256, 0, stream>>>(
      attnb, wob, bo, x, y1, nullptr, D_MODEL, D_MODEL, 0);

  // 5) LayerNorm1 -> x1 (f32) + x1b (bf16)
  layernorm<<<MROWS, 256, 0, stream>>>(y1, a1, b1, x1, x1b);

  // 6) FFN up + ReLU -> hb (bf16)
  gemm_bf16<0><<<dim3(FFH / 64, MROWS / 64), 256, 0, stream>>>(
      x1b, w1b, c1, nullptr, nullptr, hb, FFH, D_MODEL, 1);

  // 7) FFN down + residual(x1) -> y2 (f32)
  gemm_bf16<0><<<dim3(D_MODEL / 64, MROWS / 64), 256, 0, stream>>>(
      hb, w2b, c2, x1, y2, nullptr, D_MODEL, FFH, 0);

  // 8) LayerNorm2 -> final output
  layernorm<<<MROWS, 256, 0, stream>>>(y2, a2, b2, (float*)d_out, nullptr);

  (void)in_sizes; (void)n_in; (void)out_size; (void)ws_size;
}
